// TimeMixing_49057116455666
// MI455X (gfx1250) — matrix operand order repaired and checked by reading, not compiled, not run on hardware
//
#include <hip/hip_runtime.h>

typedef __attribute__((ext_vector_type(16))) __bf16 v16bf;
typedef __attribute__((ext_vector_type(8)))  float  v8f;

static constexpr int   Tn = 8192;
static constexpr int   En = 1024;
static constexpr size_t TE = (size_t)Tn * En;
static constexpr int   NC = 64;        // scan chunks
static constexpr int   CL = Tn / NC;   // 128 steps per chunk

#define BM 128
#define BN 128
#define BK 32
#define LDSA 40   // padded LDS row stride (bf16 elems): 80B, 16B-aligned, conflict-free

union FragB16 { v16bf v; uint4 q[2]; };

__device__ __forceinline__ __bf16 f2bf(float f) { return (__bf16)f; }

// --- CDNA5 async copy: global -> LDS, tracked by ASYNCcnt ------------------
__device__ __forceinline__ void async_load_b128(unsigned lds_off, const void* gaddr) {
  // VDST = LDS byte address VGPR, VADDR = 64-bit global address (GV mode)
  asm volatile("global_load_async_to_lds_b128 %0, %1, off"
               :: "v"(lds_off), "v"(gaddr) : "memory");
}
__device__ __forceinline__ void wait_async_le4() {
  asm volatile("s_wait_asynccnt 0x4" ::: "memory");
}
__device__ __forceinline__ void wait_async_0() {
  asm volatile("s_wait_asynccnt 0x0" ::: "memory");
}
__device__ __forceinline__ unsigned lds_off_u32(const void* p) {
  // generic LDS address = {shared aperture, 32-bit LDS offset}; low 32 bits
  // are exactly the hardware LDS byte address (ISA 10.2 aperture rules).
  return (unsigned)(unsigned long long)p;
}

// ---------------------------------------------------------------------------
// Token-shift mix: kx/vx/rx = x*m + prev*(1-m), emitted directly in bf16.
// ---------------------------------------------------------------------------
__global__ __launch_bounds__(256) void mix_kernel(
    const float* __restrict__ x, const float* __restrict__ sx,
    const float* __restrict__ tmk, const float* __restrict__ tmv,
    const float* __restrict__ tmr,
    __bf16* __restrict__ kxB, __bf16* __restrict__ vxB, __bf16* __restrict__ rxB) {
  size_t idx = (size_t)blockIdx.x * 256 + threadIdx.x;
  int e = (int)(idx & (En - 1));
  float xv = x[idx];
  float xp = (idx < (size_t)En) ? sx[e] : x[idx - En];
  float mk = tmk[e], mv = tmv[e], mr = tmr[e];
  kxB[idx] = f2bf(xv * mk + xp * (1.f - mk));
  vxB[idx] = f2bf(xv * mv + xp * (1.f - mv));
  rxB[idx] = f2bf(xv * mr + xp * (1.f - mr));
}

// ---------------------------------------------------------------------------
// W [K,N] f32  ->  W^T [N,K] bf16 (tiled, coalesced both sides)
// ---------------------------------------------------------------------------
__global__ void transpose_bf16_kernel(const float* __restrict__ W,
                                      __bf16* __restrict__ WT) {
  __shared__ float tile[32][33];
  int bx = blockIdx.x * 32, by = blockIdx.y * 32;
  int tx = threadIdx.x, ty = threadIdx.y;   // block (32, 8)
  #pragma unroll
  for (int i = 0; i < 32; i += 8)
    tile[ty + i][tx] = W[(size_t)(by + ty + i) * En + bx + tx];
  __syncthreads();
  #pragma unroll
  for (int i = 0; i < 32; i += 8)
    WT[(size_t)(bx + ty + i) * En + by + tx] = f2bf(tile[tx][ty + i]);
}

// ---------------------------------------------------------------------------
// C[M,N] = A[M,K](bf16) @ BT[N,K]^T(bf16), f32 accumulate via WMMA.
// Async double-buffered LDS staging (global_load_async_to_lds_b128).
// 256 threads = 8 waves; block tile 128x128x32; wave tile 32x64 (2x4 WMMAs).
// ---------------------------------------------------------------------------
__global__ __launch_bounds__(256) void gemm_bf16_wmma_kernel(
    const __bf16* __restrict__ A, const __bf16* __restrict__ BT,
    float* __restrict__ C, const float* __restrict__ addSrc,
    int M, int N, int K, int applySigmoid) {
  __shared__ __align__(16) __bf16 As[2][BM * LDSA];
  __shared__ __align__(16) __bf16 Bs[2][BN * LDSA];

  int tid  = threadIdx.x;
  int bm   = blockIdx.y * BM;
  int bn   = blockIdx.x * BN;
  int w    = tid >> 5, lane = tid & 31;
  int wm   = (w >> 1) * 32;   // 0,32,64,96
  int wn   = (w & 1) * 64;    // 0,64
  int lm   = lane & 15;
  int hi   = lane >> 4;       // half-wave select

  v8f acc[2][4];
  #pragma unroll
  for (int i = 0; i < 2; i++)
    #pragma unroll
    for (int j = 0; j < 4; j++)
      acc[i][j] = (v8f){0.f,0.f,0.f,0.f,0.f,0.f,0.f,0.f};

  int r0 = tid >> 2;          // 0..63
  int c0 = (tid & 3) * 8;     // 0,8,16,24 (bf16 elems)

  // per-thread LDS destinations (byte offsets) for the 4 async b128 copies
  unsigned la0[2], la1[2], lb0[2], lb1[2];
  #pragma unroll
  for (int b = 0; b < 2; b++) {
    la0[b] = lds_off_u32(&As[b][(size_t)r0        * LDSA + c0]);
    la1[b] = lds_off_u32(&As[b][(size_t)(r0 + 64) * LDSA + c0]);
    lb0[b] = lds_off_u32(&Bs[b][(size_t)r0        * LDSA + c0]);
    lb1[b] = lds_off_u32(&Bs[b][(size_t)(r0 + 64) * LDSA + c0]);
  }
  const __bf16* gA0 = A  + (size_t)(bm + r0)      * K + c0;
  const __bf16* gA1 = A  + (size_t)(bm + r0 + 64) * K + c0;
  const __bf16* gB0 = BT + (size_t)(bn + r0)      * K + c0;
  const __bf16* gB1 = BT + (size_t)(bn + r0 + 64) * K + c0;

  // prologue: tile 0 into buffer 0
  async_load_b128(la0[0], gA0);
  async_load_b128(la1[0], gA1);
  async_load_b128(lb0[0], gB0);
  async_load_b128(lb1[0], gB1);

  for (int kt = 0; kt < K; kt += BK) {
    int p = (kt >> 5) & 1;
    bool more = (kt + BK) < K;
    if (more) {           // issue next tile into the other buffer
      int nb = 1 - p;
      async_load_b128(la0[nb], gA0 + kt + BK);
      async_load_b128(la1[nb], gA1 + kt + BK);
      async_load_b128(lb0[nb], gB0 + kt + BK);
      async_load_b128(lb1[nb], gB1 + kt + BK);
      wait_async_le4();   // async loads complete in order -> current tile landed
    } else {
      wait_async_0();
    }
    __syncthreads();

    FragB16 af[2], bfr[4];
    // A 16x32 bf16 layout: lane<16 -> K 0..7 & 16..23 ; lane>=16 -> K 8..15 & 24..31
    #pragma unroll
    for (int i = 0; i < 2; i++) {
      const __bf16* fp = &As[p][(size_t)(wm + i * 16 + lm) * LDSA + hi * 8];
      af[i].q[0] = *(const uint4*)(fp);
      af[i].q[1] = *(const uint4*)(fp + 16);
    }
    // B 32x16 bf16 layout: lane<16 -> K 0..7 & 16..23 ; lane>=16 -> K 8..15 & 24..31 (col = lm)
    #pragma unroll
    for (int j = 0; j < 4; j++) {
      const __bf16* fp = &Bs[p][(size_t)(wn + j * 16 + lm) * LDSA + hi * 8];
      bfr[j].q[0] = *(const uint4*)(fp);
      bfr[j].q[1] = *(const uint4*)(fp + 16);
    }
    #pragma unroll
    for (int i = 0; i < 2; i++)
      #pragma unroll
      for (int j = 0; j < 4; j++)
        acc[i][j] = __builtin_amdgcn_wmma_f32_16x16x32_bf16(
            false, af[i].v, false, bfr[j].v, (short)0, acc[i][j], false, false);

    __syncthreads();      // WAR: next iteration's async copies reuse 1-p buffer
  }

  // C/D layout: VGPR v, lane -> row = v + hi*8, col = lm
  #pragma unroll
  for (int i = 0; i < 2; i++) {
    #pragma unroll
    for (int j = 0; j < 4; j++) {
      int gc = bn + wn + j * 16 + lm;
      #pragma unroll
      for (int vreg = 0; vreg < 8; vreg++) {
        int gr = bm + wm + i * 16 + hi * 8 + vreg;
        float val = acc[i][j][vreg];
        if (applySigmoid) val = 1.0f / (1.0f + __expf(-val));
        if (addSrc) val += addSrc[(size_t)gr * N + gc];
        C[(size_t)gr * N + gc] = val;
      }
    }
  }
}

// ---------------------------------------------------------------------------
// WKV scan, pass 1: per-chunk transition summary starting from zero state
// (aa=bb=0, pp=-inf). The recurrence is a stabilized linear recurrence, so
// chunk summaries compose associatively with single-step semantics.
// ---------------------------------------------------------------------------
__global__ __launch_bounds__(256) void wkv_chunk_local_kernel(
    const float* __restrict__ kF, const float* __restrict__ vF,
    const float* __restrict__ time_decay,
    float* __restrict__ cAA, float* __restrict__ cBB, float* __restrict__ cPP) {
  int g = blockIdx.x * 256 + threadIdx.x;   // NC*E threads
  int e = g & (En - 1);
  int c = g >> 10;
  float dec = -__expf(time_decay[e]);
  float aa = 0.f, bb = 0.f, pp = -1e30f;
  int t0 = c * CL;
  for (int t = 0; t < CL; t++) {
    size_t i = (size_t)(t0 + t) * En + e;
    float kk = kF[i], vv = vF[i];
    float ww2 = dec + pp;
    float p2  = fmaxf(ww2, kk);
    float e1  = __expf(ww2 - p2);
    float e2  = __expf(kk - p2);
    aa = e1 * aa + e2 * vv;
    bb = e1 * bb + e2;
    pp = p2;
  }
  size_t o = (size_t)c * En + e;
  cAA[o] = aa; cBB[o] = bb; cPP[o] = pp;
}

// ---------------------------------------------------------------------------
// Pass 2: serial prefix over the 64 chunk summaries (per channel) ->
// incoming state for each chunk; also emits final (x_last, aa, bb, pp).
// ---------------------------------------------------------------------------
__global__ __launch_bounds__(256) void wkv_chunk_prefix_kernel(
    const float* __restrict__ aa_in, const float* __restrict__ bb_in,
    const float* __restrict__ pp_in, const float* __restrict__ time_decay,
    const float* __restrict__ cAA, const float* __restrict__ cBB,
    const float* __restrict__ cPP,
    float* __restrict__ sAA, float* __restrict__ sBB, float* __restrict__ sPP,
    const float* __restrict__ x, float* __restrict__ outbuf) {
  int e = blockIdx.x * 256 + threadIdx.x;
  float aa = aa_in[e], bb = bb_in[e], pp = pp_in[e];
  float D = (float)CL * (-__expf(time_decay[e]));   // chunk total decay (log)
  for (int c = 0; c < NC; c++) {
    size_t i = (size_t)c * En + e;
    sAA[i] = aa; sBB[i] = bb; sPP[i] = pp;
    float w  = pp + D;
    float pc = cPP[i];
    float p  = fmaxf(w, pc);
    float e1 = __expf(w - p);
    float e2 = __expf(pc - p);
    aa = e1 * aa + e2 * cAA[i];
    bb = e1 * bb + e2 * cBB[i];
    pp = p;
  }
  size_t base = TE;
  outbuf[base + e]          = x[(size_t)(Tn - 1) * En + e];  // x_last
  outbuf[base + En + e]     = aa;                            // aa_f
  outbuf[base + 2 * En + e] = bb;                            // bb_f
  outbuf[base + 3 * En + e] = pp;                            // pp_f
}

// ---------------------------------------------------------------------------
// Pass 3: per-chunk output emission from the incoming state; writes
// r*wkv directly in bf16 for the output GEMM.
// ---------------------------------------------------------------------------
__global__ __launch_bounds__(256) void wkv_chunk_out_kernel(
    const float* __restrict__ kF, const float* __restrict__ vF,
    const float* __restrict__ rF,
    const float* __restrict__ sAA, const float* __restrict__ sBB,
    const float* __restrict__ sPP,
    const float* __restrict__ time_first, const float* __restrict__ time_decay,
    __bf16* __restrict__ rwkvB) {
  int g = blockIdx.x * 256 + threadIdx.x;   // NC*E threads
  int e = g & (En - 1);
  int c = g >> 10;
  size_t o = (size_t)c * En + e;
  float aa = sAA[o], bb = sBB[o], pp = sPP[o];
  float tf  = time_first[e];
  float dec = -__expf(time_decay[e]);
  int t0 = c * CL;
  for (int t = 0; t < CL; t++) {
    size_t i = (size_t)(t0 + t) * En + e;
    float kk = kF[i], vv = vF[i], rr = rF[i];
    float ww = tf + kk;
    float p  = fmaxf(pp, ww);
    float e1 = __expf(pp - p);
    float e2 = __expf(ww - p);
    float wkv = (e1 * aa + e2 * vv) / (e1 * bb + e2);
    rwkvB[i] = f2bf(rr * wkv);
    float ww2 = dec + pp;
    float p2  = fmaxf(ww2, kk);
    float e1b = __expf(ww2 - p2);
    float e2b = __expf(kk - p2);
    aa = e1b * aa + e2b * vv;
    bb = e1b * bb + e2b;
    pp = p2;
  }
}

// ---------------------------------------------------------------------------
extern "C" void kernel_launch(void* const* d_in, const int* in_sizes, int n_in,
                              void* d_out, int out_size, void* d_ws, size_t ws_size,
                              hipStream_t stream) {
  const float* x   = (const float*)d_in[0];
  const float* sx  = (const float*)d_in[1];
  const float* aa0 = (const float*)d_in[2];
  const float* bb0 = (const float*)d_in[3];
  const float* pp0 = (const float*)d_in[4];
  const float* tf  = (const float*)d_in[5];
  const float* td  = (const float*)d_in[6];
  const float* tmk = (const float*)d_in[7];
  const float* tmv = (const float*)d_in[8];
  const float* tmr = (const float*)d_in[9];
  const float* Wk  = (const float*)d_in[10];
  const float* Wv  = (const float*)d_in[11];
  const float* Wr  = (const float*)d_in[12];
  const float* Wo  = (const float*)d_in[13];
  float* out = (float*)d_out;

  // workspace carve-up
  char* ws = (char*)d_ws;
  __bf16* kxB   = (__bf16*)ws;  ws += TE * 2;
  __bf16* vxB   = (__bf16*)ws;  ws += TE * 2;
  __bf16* rxB   = (__bf16*)ws;  ws += TE * 2;
  __bf16* rwkvB = (__bf16*)ws;  ws += TE * 2;
  __bf16* WkT   = (__bf16*)ws;  ws += (size_t)En * En * 2;
  __bf16* WvT   = (__bf16*)ws;  ws += (size_t)En * En * 2;
  __bf16* WrT   = (__bf16*)ws;  ws += (size_t)En * En * 2;
  __bf16* WoT   = (__bf16*)ws;  ws += (size_t)En * En * 2;
  float* kF = (float*)ws;  ws += TE * 4;
  float* vF = (float*)ws;  ws += TE * 4;
  float* rF = (float*)ws;  ws += TE * 4;
  float* cAA = (float*)ws; ws += (size_t)NC * En * 4;
  float* cBB = (float*)ws; ws += (size_t)NC * En * 4;
  float* cPP = (float*)ws; ws += (size_t)NC * En * 4;
  float* sAA = (float*)ws; ws += (size_t)NC * En * 4;
  float* sBB = (float*)ws; ws += (size_t)NC * En * 4;
  float* sPP = (float*)ws; ws += (size_t)NC * En * 4;

  // 1) token-shift mixes (f32 -> bf16)
  mix_kernel<<<(int)(TE / 256), 256, 0, stream>>>(x, sx, tmk, tmv, tmr,
                                                  kxB, vxB, rxB);
  // 2) weight transposes (f32 [K,N] -> bf16 [N,K])
  dim3 tb(32, 8), tg(En / 32, En / 32);
  transpose_bf16_kernel<<<tg, tb, 0, stream>>>(Wk, WkT);
  transpose_bf16_kernel<<<tg, tb, 0, stream>>>(Wv, WvT);
  transpose_bf16_kernel<<<tg, tb, 0, stream>>>(Wr, WrT);
  transpose_bf16_kernel<<<tg, tb, 0, stream>>>(Wo, WoT);

  // 3) k / v / r GEMMs (sigmoid fused into r)
  dim3 gg(En / BN, Tn / BM);
  gemm_bf16_wmma_kernel<<<gg, 256, 0, stream>>>(kxB, WkT, kF, nullptr,
                                                Tn, En, En, 0);
  gemm_bf16_wmma_kernel<<<gg, 256, 0, stream>>>(vxB, WvT, vF, nullptr,
                                                Tn, En, En, 0);
  gemm_bf16_wmma_kernel<<<gg, 256, 0, stream>>>(rxB, WrT, rF, nullptr,
                                                Tn, En, En, 1);

  // 4) chunked WKV scan (parallel local -> serial chunk prefix -> parallel emit)
  wkv_chunk_local_kernel<<<(NC * En) / 256, 256, 0, stream>>>(kF, vF, td,
                                                              cAA, cBB, cPP);
  wkv_chunk_prefix_kernel<<<En / 256, 256, 0, stream>>>(aa0, bb0, pp0, td,
                                                        cAA, cBB, cPP,
                                                        sAA, sBB, sPP, x, out);
  wkv_chunk_out_kernel<<<(NC * En) / 256, 256, 0, stream>>>(kF, vF, rF,
                                                            sAA, sBB, sPP,
                                                            tf, td, rwkvB);

  // 5) out = x + (r*wkv) @ Wo  (residual fused)
  gemm_bf16_wmma_kernel<<<gg, 256, 0, stream>>>(rwkvB, WoT, out, x,
                                                Tn, En, En, 0);
}
